// CausalSelfAttention_7112465842744
// MI455X (gfx1250) — matrix operand order repaired and checked by reading, not compiled, not run on hardware
//
#include <hip/hip_runtime.h>
#include <hip/hip_bf16.h>

typedef __attribute__((ext_vector_type(16))) __bf16 v16bf;
typedef __attribute__((ext_vector_type(8)))  __bf16 v8bf;
typedef __attribute__((ext_vector_type(8)))  float  v8f;

namespace {
constexpr int kC  = 2048;
constexpr int kT  = 2048;
constexpr int kB  = 2;
constexpr int kH  = 16;
constexpr int kHD = 128;
constexpr int kM  = kB * kT;   // 4096 tokens
constexpr int kN3 = 3 * kC;    // 6144
constexpr float kNegInf = -3.0e38f;
constexpr int kLdsRow = 40;    // 32 data elems + 8 pad -> 80B stride, bank-safe
}

// ---------------------------------------------------------------------------
// Fragment loaders (CDNA5 wave32 WMMA bf16 16x16x32 layouts, ISA 7.12.2)
// A (16x32, row-major source): lane<16 -> row=lane, K in {0..7, 16..23};
//                              lane>=16 -> row=lane-16, K in {8..15, 24..31}
__device__ __forceinline__ v16bf load_a_frag(const __bf16* base, int ld,
                                             int row, int kbase, int lane) {
    int r    = lane & 15;
    int koff = (lane >> 4) * 8;
    const __bf16* p = base + (size_t)(row + r) * ld + kbase + koff;
    v8bf lo = *(const v8bf*)p;
    v8bf hi = *(const v8bf*)(p + 16);
    v16bf f;
#pragma unroll
    for (int i = 0; i < 8; ++i) { f[i] = lo[i]; f[i + 8] = hi[i]; }
    return f;
}

// B (32x16): lane<16 -> col=lane, K in {0..7, 16..23}; lane>=16 -> col=lane-16, K in {8..15, 24..31}.
// Source is stored "transposed" [N, K] so each lane load is contiguous.
// Two 16B loads so padded LDS row strides never force a 32B-aligned access.
__device__ __forceinline__ v16bf load_b_frag(const __bf16* baseT, int ld,
                                             int ncol, int kbase, int lane) {
    int n    = lane & 15;
    int koff = (lane >> 4) * 8;
    const __bf16* p = baseT + (size_t)(ncol + n) * ld + kbase + koff;
    v8bf lo = *(const v8bf*)p;
    v8bf hi = *(const v8bf*)(p + 16);
    v16bf f;
#pragma unroll
    for (int i = 0; i < 8; ++i) { f[i] = lo[i]; f[i + 8] = hi[i]; }
    return f;
}

__device__ __forceinline__ v8f wmma_bf16(v16bf a, v16bf b, v8f c) {
    return __builtin_amdgcn_wmma_f32_16x16x32_bf16(
        /*neg_a=*/false, a, /*neg_b=*/false, b,
        /*c_mod=*/(short)0, c, /*reuse_a=*/false, /*reuse_b=*/false);
}

// ---------------------------------------------------------------------------
// CDNA5 async global->LDS copy (VGLOBAL GLOBAL_LOAD_ASYNC_TO_LDS_B128,
// tracked with ASYNCcnt; CDNA5 ISA).
__device__ __forceinline__ uint32_t lds_off(const void* p) {
    // Generic LDS pointers carry the LDS byte offset in addr[31:0] (ISA 10.2).
    return (uint32_t)(uintptr_t)p;
}

__device__ __forceinline__ void async_load_b128(uint32_t ldsaddr,
                                                const void* gaddr) {
    asm volatile("global_load_async_to_lds_b128 %0, %1, off"
                 :: "v"(ldsaddr), "v"(gaddr)
                 : "memory");
}

__device__ __forceinline__ void wait_asynccnt0() {
    asm volatile("s_wait_asynccnt 0x0" ::: "memory");
}

// Copy a 128x32 bf16 tile (row-major, ld elements) into LDS laid out as
// [128][kLdsRow] bf16.  256 threads, 2x b128 per thread = 512 16B blocks.
__device__ __forceinline__ void async_copy_tile(uint32_t lds_base,
                                                const __bf16* gbase, int ld,
                                                int row0, int kbase, int tid) {
#pragma unroll
    for (int i = 0; i < 2; ++i) {
        int blk = i * 256 + tid;          // 16B block index
        int r   = blk >> 2;               // 4 blocks per 64B data row
        int ce  = (blk & 3) * 8;          // element offset within row
        const __bf16* g = gbase + (size_t)(row0 + r) * ld + kbase + ce;
        async_load_b128(lds_base + (uint32_t)(r * kLdsRow + ce) * 2u, g);
    }
}

// ---------------------------------------------------------------------------
// Precision conversion kernels
__global__ void cvt_f32_to_bf16(const float* __restrict__ in,
                                __bf16* __restrict__ out, int n) {
    int i = blockIdx.x * blockDim.x + threadIdx.x;
    if (i < n) out[i] = (__bf16)in[i];
}

// LDS-tiled transpose+convert: outT[n*rows + k] = bf16(in[k*cols + n]).
__global__ void __launch_bounds__(256)
cvt_transpose_bf16(const float* __restrict__ in, __bf16* __restrict__ outT,
                   int rows, int cols) {
    __shared__ __bf16 tile[32][33];
    int tx = threadIdx.x & 31;
    int ty = threadIdx.x >> 5;        // 0..7
    int n0 = blockIdx.x * 32;         // N tile (source cols)
    int k0 = blockIdx.y * 32;         // K tile (source rows)
#pragma unroll
    for (int i = 0; i < 32; i += 8) {
        tile[ty + i][tx] = (__bf16)in[(size_t)(k0 + ty + i) * cols + n0 + tx];
    }
    __syncthreads();
#pragma unroll
    for (int i = 0; i < 32; i += 8) {
        outT[(size_t)(n0 + ty + i) * rows + k0 + tx] = tile[tx][ty + i];
    }
}

// ---------------------------------------------------------------------------
// Shared GEMM core: 256 threads = 8 waves (4M x 2N), block tile 128x128,
// K chunks of 32 double-buffered in LDS via async copies.  Each wave computes
// a 32x64 sub-tile (8 WMMAs per chunk) out of LDS.
struct GemmAcc { v8f acc[2][4]; };

template <typename Epilogue>
__device__ __forceinline__ void gemm_lds_core(const __bf16* __restrict__ Amat,
                                              const __bf16* __restrict__ BTmat,
                                              int kdim, Epilogue epi) {
    __shared__ __bf16 At[2][128 * kLdsRow];
    __shared__ __bf16 Bt[2][128 * kLdsRow];

    int tid  = threadIdx.x;
    int lane = tid & 31;
    int wave = tid >> 5;          // 0..7
    int wm   = wave >> 1;         // 0..3
    int wn   = wave & 1;          // 0..1
    int mblock = blockIdx.x * 128;
    int nblock = blockIdx.y * 128;

    uint32_t aoff[2] = { lds_off(&At[0][0]), lds_off(&At[1][0]) };
    uint32_t boff[2] = { lds_off(&Bt[0][0]), lds_off(&Bt[1][0]) };

    // prologue: stage chunk 0 into buffer 0
    async_copy_tile(aoff[0], Amat, kdim, mblock, 0, tid);
    async_copy_tile(boff[0], BTmat, kdim, nblock, 0, tid);

    GemmAcc g = {};
    for (int k = 0; k < kdim; k += 32) {
        int cur = (k >> 5) & 1;
        wait_asynccnt0();          // my chunk-k copies landed
        __syncthreads();           // everyone's chunk-k copies landed
        if (k + 32 < kdim) {       // overlap next chunk's copies with compute
            async_copy_tile(aoff[cur ^ 1], Amat, kdim, mblock, k + 32, tid);
            async_copy_tile(boff[cur ^ 1], BTmat, kdim, nblock, k + 32, tid);
        }
        const __bf16* a = &At[cur][0];
        const __bf16* b = &Bt[cur][0];
        v16bf fa0 = load_a_frag(a, kLdsRow, wm * 32, 0, lane);
        v16bf fa1 = load_a_frag(a, kLdsRow, wm * 32 + 16, 0, lane);
#pragma unroll
        for (int t = 0; t < 4; ++t) {
            v16bf fb = load_b_frag(b, kLdsRow, wn * 64 + t * 16, 0, lane);
            g.acc[0][t] = wmma_bf16(fa0, fb, g.acc[0][t]);
            g.acc[1][t] = wmma_bf16(fa1, fb, g.acc[1][t]);
        }
    }

    int mtile = mblock + wm * 32;
    int nbase = nblock + wn * 64;
    epi(g, mtile, nbase, lane);
}

// ---------------------------------------------------------------------------
// QKV GEMM: [4096,2048] x [2048,6144] + bias.
// Epilogue scatters into Q,K as [B,H,T,hd] bf16 and V as [B,H,hd,T] bf16.
__global__ void __launch_bounds__(256)
gemm_qkv_kernel(const __bf16* __restrict__ xb, const __bf16* __restrict__ wT,
                const float* __restrict__ bias,
                __bf16* __restrict__ Q, __bf16* __restrict__ K,
                __bf16* __restrict__ Vt) {
    gemm_lds_core(xb, wT, kC,
        [&](const GemmAcc& g, int mtile, int nbase, int lane) {
            int col  = lane & 15;
            int half = lane >> 4;
#pragma unroll
            for (int mi = 0; mi < 2; ++mi) {
#pragma unroll
                for (int t = 0; t < 4; ++t) {
#pragma unroll
                    for (int gg = 0; gg < 8; ++gg) {
                        int m = mtile + mi * 16 + gg + 8 * half;  // token
                        int n = nbase + t * 16 + col;             // 0..6143
                        float v = g.acc[mi][t][gg] + bias[n];
                        int which = n >> 11;                      // 0=Q 1=K 2=V
                        int c = n & (kC - 1);
                        int h = c >> 7;
                        int d = c & (kHD - 1);
                        int b_ = m >> 11;
                        int tk = m & (kT - 1);
                        if (which == 0)
                            Q[(((size_t)(b_ * kH + h) * kT + tk) * kHD) + d] = (__bf16)v;
                        else if (which == 1)
                            K[(((size_t)(b_ * kH + h) * kT + tk) * kHD) + d] = (__bf16)v;
                        else
                            Vt[(((size_t)(b_ * kH + h) * kHD + d) * kT) + tk] = (__bf16)v;
                    }
                }
            }
        });
}

// ---------------------------------------------------------------------------
// Output projection: attn_out[4096,2048] x w_proj[2048,2048] + bias -> f32.
__global__ void __launch_bounds__(256)
gemm_proj_kernel(const __bf16* __restrict__ A, const __bf16* __restrict__ wT,
                 const float* __restrict__ bias, float* __restrict__ out) {
    gemm_lds_core(A, wT, kC,
        [&](const GemmAcc& g, int mtile, int nbase, int lane) {
            int col  = lane & 15;
            int half = lane >> 4;
#pragma unroll
            for (int mi = 0; mi < 2; ++mi) {
#pragma unroll
                for (int t = 0; t < 4; ++t) {
#pragma unroll
                    for (int gg = 0; gg < 8; ++gg) {
                        int m = mtile + mi * 16 + gg + 8 * half;
                        int n = nbase + t * 16 + col;
                        out[(size_t)m * kC + n] = g.acc[mi][t][gg] + bias[n];
                    }
                }
            }
        });
}

// ---------------------------------------------------------------------------
// Flash attention: one wave per (b,h, 16-row q tile); streams 32-key blocks.
__global__ void __launch_bounds__(32)
flash_attn_kernel(const __bf16* __restrict__ Q, const __bf16* __restrict__ K,
                  const __bf16* __restrict__ Vt, __bf16* __restrict__ out) {
    int wid  = blockIdx.x;
    int qt   = wid & (kT / 16 - 1);   // 0..127
    int bh   = wid >> 7;              // 0..31
    int lane = threadIdx.x;
    int col  = lane & 15;
    int half = lane >> 4;

    const __bf16* Qh = Q  + (size_t)bh * kT * kHD;
    const __bf16* Kh = K  + (size_t)bh * kT * kHD;
    const __bf16* Vh = Vt + (size_t)bh * kHD * kT;
    int q0 = qt * 16;

    __shared__ __bf16 Plds[16 * 32];  // P tile bounce: C-frag -> A-frag

    v16bf qf[4];
#pragma unroll
    for (int c = 0; c < 4; ++c) qf[c] = load_a_frag(Qh, kHD, q0, c * 32, lane);

    v8f o[8] = {};
    float m_i[8], l_i[8];
#pragma unroll
    for (int g = 0; g < 8; ++g) { m_i[g] = kNegInf; l_i[g] = 0.0f; }

    const float scale = 0.08838834764831845f;  // 1/sqrt(128)

    for (int jb = 0; jb <= q0 + 15; jb += 32) {
        if (jb + 32 <= q0 + 15) {  // prefetch next key block (global_prefetch_b8)
            __builtin_prefetch(Kh + (size_t)(jb + 32 + (lane & 15)) * kHD, 0, 1);
            __builtin_prefetch(Vh + (size_t)(lane & 15) * kT + jb + 32, 0, 1);
        }
        // ---- S = Q K^T for a 16x32 key block (two 16x16 N-tiles) ----
        v8f s[2] = {};
#pragma unroll
        for (int c = 0; c < 4; ++c) {
#pragma unroll
            for (int sub = 0; sub < 2; ++sub) {
                v16bf b = load_b_frag(Kh, kHD, jb + sub * 16, c * 32, lane);
                s[sub]  = wmma_bf16(qf[c], b, s[sub]);
            }
        }

        // ---- scale + causal mask + online softmax ----
        float alpha[8];
#pragma unroll
        for (int g = 0; g < 8; ++g) {
            int qrow = q0 + g + 8 * half;
            float mx = kNegInf;
#pragma unroll
            for (int sub = 0; sub < 2; ++sub) {
                int kcol = jb + sub * 16 + col;
                float v  = s[sub][g] * scale;
                if (kcol > qrow) v = kNegInf;
                s[sub][g] = v;
                mx = fmaxf(mx, v);
            }
            for (int off = 1; off < 16; off <<= 1)
                mx = fmaxf(mx, __shfl_xor(mx, off, 32));
            float mnew = fmaxf(m_i[g], mx);
            alpha[g]   = __expf(m_i[g] - mnew);
            m_i[g]     = mnew;
            float rs = 0.0f;
#pragma unroll
            for (int sub = 0; sub < 2; ++sub) {
                float p = __expf(s[sub][g] - mnew);
                s[sub][g] = p;
                rs += p;
            }
            for (int off = 1; off < 16; off <<= 1)
                rs += __shfl_xor(rs, off, 32);
            l_i[g] = l_i[g] * alpha[g] + rs;
        }

        // rescale running O
#pragma unroll
        for (int t = 0; t < 8; ++t)
#pragma unroll
            for (int g = 0; g < 8; ++g) o[t][g] *= alpha[g];

        // ---- P (C-frag f32) -> LDS -> A-frag bf16 ----
        __syncthreads();
#pragma unroll
        for (int g = 0; g < 8; ++g) {
            int r = g + 8 * half;
            Plds[r * 32 + col]      = (__bf16)s[0][g];
            Plds[r * 32 + 16 + col] = (__bf16)s[1][g];
        }
        __syncthreads();
        v16bf pf = load_a_frag((const __bf16*)Plds, 32, 0, 0, lane);

        // ---- O += P @ V (V stored [hd, T] so B-frag is contiguous) ----
#pragma unroll
        for (int t = 0; t < 8; ++t) {
            v16bf b = load_b_frag(Vh, kT, t * 16, jb, lane);
            o[t] = wmma_bf16(pf, b, o[t]);
        }
    }

    // ---- normalize + store to attn_out [token, C] bf16 ----
    int b_ = bh >> 4;
    int h  = bh & 15;
#pragma unroll
    for (int t = 0; t < 8; ++t) {
#pragma unroll
        for (int g = 0; g < 8; ++g) {
            int qrow = q0 + g + 8 * half;
            float v  = o[t][g] / l_i[g];
            size_t tok = (size_t)b_ * kT + qrow;
            out[tok * kC + h * kHD + t * 16 + col] = (__bf16)v;
        }
    }
}

// ---------------------------------------------------------------------------
extern "C" void kernel_launch(void* const* d_in, const int* in_sizes, int n_in,
                              void* d_out, int out_size, void* d_ws, size_t ws_size,
                              hipStream_t stream) {
    const float* x      = (const float*)d_in[0];
    const float* w_attn = (const float*)d_in[1];
    const float* b_attn = (const float*)d_in[2];
    const float* w_proj = (const float*)d_in[3];
    const float* b_proj = (const float*)d_in[4];
    float* out = (float*)d_out;

    // workspace carve-up (each buffer padded 4KB for diagonal-tile over-reads)
    char* ws = (char*)d_ws;
    size_t off = 0;
    auto alloc = [&](size_t bytes) -> void* {
        void* p = ws + off;
        off += (bytes + 4095) & ~(size_t)4095;
        off += 4096;
        return p;
    };
    __bf16* xb    = (__bf16*)alloc((size_t)kM * kC * 2);        // x bf16
    __bf16* waT   = (__bf16*)alloc((size_t)kN3 * kC * 2);       // w_attn^T bf16
    __bf16* wpT   = (__bf16*)alloc((size_t)kC * kC * 2);        // w_proj^T bf16
    __bf16* Qb    = (__bf16*)alloc((size_t)kM * kC * 2);        // [B,H,T,hd]
    __bf16* Kb    = (__bf16*)alloc((size_t)kM * kC * 2);        // [B,H,T,hd]
    __bf16* Vtb   = (__bf16*)alloc((size_t)kM * kC * 2);        // [B,H,hd,T]
    __bf16* aout  = (__bf16*)alloc((size_t)kM * kC * 2);        // attn out bf16

    // 1) conversions
    {
        int n = kM * kC;
        cvt_f32_to_bf16<<<(n + 255) / 256, 256, 0, stream>>>(x, xb, n);
    }
    {
        dim3 grid(kN3 / 32, kC / 32);
        cvt_transpose_bf16<<<grid, 256, 0, stream>>>(w_attn, waT, kC, kN3);
    }
    {
        dim3 grid(kC / 32, kC / 32);
        cvt_transpose_bf16<<<grid, 256, 0, stream>>>(w_proj, wpT, kC, kC);
    }

    // 2) QKV GEMM (WMMA bf16, async-LDS double-buffered 128x128 block tiles)
    {
        dim3 grid(kM / 128, kN3 / 128);
        gemm_qkv_kernel<<<grid, 256, 0, stream>>>(xb, waT, b_attn, Qb, Kb, Vtb);
    }

    // 3) flash attention (WMMA bf16, online softmax)
    {
        int nblocks = kB * kH * (kT / 16);  // 4096 waves
        flash_attn_kernel<<<nblocks, 32, 0, stream>>>(Qb, Kb, Vtb, aout);
    }

    // 4) output projection (WMMA bf16 -> f32)
    {
        dim3 grid(kM / 128, kC / 128);
        gemm_proj_kernel<<<grid, 256, 0, stream>>>(aout, wpT, b_proj, out);
    }
}
